// OsrSAF_TriNet_82910048682287
// MI455X (gfx1250) — matrix operand order repaired and checked by reading, not compiled, not run on hardware
//
#include <hip/hip_runtime.h>
#include <stdint.h>

// ---------------------------------------------------------------------------
// Cosine distance to nearest centroid of each class.
//   codes:     (B, 256) fp32, rows already L2-normalized
//   centroids: (10, 4, 256) fp32, unnormalized
//   out:       (B, 10) fp32 = 1 - max_k <code, cent_normed[c,k]>
// B-matrix = 40 normalized centroid rows (padded to 48), bf16, pre-swizzled
// into v_wmma_f32_16x16x32_bf16 B-fragment layout and staged in LDS; codes
// streamed once from HBM with non-temporal hints (1 GiB stream >> 192 MB L2;
// memory-bound floor ~46us at 23.3 TB/s).
// ---------------------------------------------------------------------------

typedef __attribute__((ext_vector_type(16))) __bf16 v16bf;
typedef __attribute__((ext_vector_type(4)))  __bf16 v4bf;
typedef __attribute__((ext_vector_type(8)))  float  v8f;
typedef __attribute__((ext_vector_type(4)))  float  v4f;

#define NUM_CLASSES 10
#define CROWS       40      // 10 classes * 4 centroids
#define CODE_DIM    256
#define NTILES      3       // ceil(40/16) N-tiles of 16 columns
#define KSTEPS      8       // 256 / 32
#define FRAG_ELEMS  (NTILES * KSTEPS * 32 * 16)   // bf16 elements in all B frags

// --------------------------- centroid prep ---------------------------------
// Normalize 40 rows, convert to bf16, emit B fragments in WMMA lane order:
// fragment f = nt*8 + kt holds 32 lanes x 16 bf16; lane l supplies column
// n = nt*16 + (l&15); lanes 0-15 carry K = kt*32 + {0..7,16..23}, lanes 16-31 carry
// K = kt*32 + {8..15,24..31} (matches A fragment K order). n >= 40 -> zeros.
__global__ void prep_centroid_frags(const float* __restrict__ cents,
                                    __bf16* __restrict__ frags) {
    __shared__ float rnorm[CROWS];
    int t = threadIdx.x;
    if (t < CROWS) {
        const float* row = cents + t * CODE_DIM;
        float s = 0.f;
        for (int i = 0; i < CODE_DIM; ++i) { float v = row[i]; s += v * v; }
        rnorm[t] = 1.0f / fmaxf(sqrtf(s), 1e-12f);
    }
    __syncthreads();
    for (int e = t; e < FRAG_ELEMS; e += blockDim.x) {
        int j    = e & 15;           // element within lane's v16bf
        int lane = (e >> 4) & 31;
        int f    = e >> 9;           // fragment index = nt*8 + kt
        int kt   = f & 7;
        int nt   = f >> 3;
        int n = nt * 16 + (lane & 15);
        int k = kt * 32 + ((j < 8) ? (lane >> 4) * 8 + j : 16 + (lane >> 4) * 8 + (j - 8));
        float v = 0.f;
        if (n < CROWS) v = cents[n * CODE_DIM + k] * rnorm[n];
        frags[e] = (__bf16)v;        // native RNE truncation
    }
}

// ------------------------------ main GEMM ----------------------------------
__global__ __launch_bounds__(256) void cosine_dist_kernel(
        const float* __restrict__ codes,
        const __bf16* __restrict__ frags,
        float* __restrict__ out) {
    // Stage all B fragments (24 KB) into LDS once per block.
    __shared__ uint4 sfrag[FRAG_ELEMS / 8];   // 1536 * 16B = 24 KB
    {
        const uint4* src = (const uint4*)frags;
        for (int i = threadIdx.x; i < FRAG_ELEMS / 8; i += blockDim.x)
            sfrag[i] = src[i];
    }
    __syncthreads();

    const int wave = threadIdx.x >> 5;
    const int lane = threadIdx.x & 31;
    const int m    = lane & 15;    // M row within 16-row tile
    const int hi   = lane >> 4;    // lane half selects K sub-range

    const long rowBase = ((long)blockIdx.x * 8 + wave) * 16;
    const float* arow = codes + (rowBase + m) * CODE_DIM;

    const v16bf* bb = (const v16bf*)sfrag;

    v8f acc0 = {}, acc1 = {}, acc2 = {};

    #pragma unroll
    for (int kt = 0; kt < KSTEPS; ++kt) {
        // A fragment (16-bit A 16x32 layout): lane<16 -> K {0..7,16..23},
        // lane>=16 -> K {8..15,24..31}, all offset by kt*32.
        // Single-pass stream: non-temporal so it doesn't churn WGP$/L2.
        const int kb = kt * 32 + hi * 8;
        v4f f0 = __builtin_nontemporal_load((const v4f*)(arow + kb));
        v4f f1 = __builtin_nontemporal_load((const v4f*)(arow + kb + 4));
        v4f f2 = __builtin_nontemporal_load((const v4f*)(arow + kb + 16));
        v4f f3 = __builtin_nontemporal_load((const v4f*)(arow + kb + 20));

        union { v16bf v; v4bf q[4]; } a;
        a.q[0] = __builtin_convertvector(f0, v4bf);   // v_cvt_pk_bf16_f32
        a.q[1] = __builtin_convertvector(f1, v4bf);
        a.q[2] = __builtin_convertvector(f2, v4bf);
        a.q[3] = __builtin_convertvector(f3, v4bf);

        v16bf b0 = bb[(0 * 8 + kt) * 32 + lane];
        v16bf b1 = bb[(1 * 8 + kt) * 32 + lane];
        v16bf b2 = bb[(2 * 8 + kt) * 32 + lane];

        acc0 = __builtin_amdgcn_wmma_f32_16x16x32_bf16(false, a.v, false, b0,
                                                       (short)0, acc0, false, false);
        acc1 = __builtin_amdgcn_wmma_f32_16x16x32_bf16(false, a.v, false, b1,
                                                       (short)0, acc1, false, false);
        acc2 = __builtin_amdgcn_wmma_f32_16x16x32_bf16(false, a.v, false, b2,
                                                       (short)0, acc2, false, false);
    }

    // C/D layout: element i of acc -> row M = i + 8*hi; column N = lane & 15.
    // Class = N/4 within each 16-column tile; max over the 4 lanes of a class
    // via xor-butterfly on lane bits 0,1.
    v8f m0 = acc0, m1 = acc1, m2 = acc2;
    #pragma unroll
    for (int i = 0; i < 8; ++i) {
        float v0 = m0[i], v1 = m1[i], v2 = m2[i];
        v0 = fmaxf(v0, __shfl_xor(v0, 1, 32));
        v0 = fmaxf(v0, __shfl_xor(v0, 2, 32));
        v1 = fmaxf(v1, __shfl_xor(v1, 1, 32));
        v1 = fmaxf(v1, __shfl_xor(v1, 2, 32));
        v2 = fmaxf(v2, __shfl_xor(v2, 1, 32));
        v2 = fmaxf(v2, __shfl_xor(v2, 2, 32));
        m0[i] = v0; m1[i] = v1; m2[i] = v2;
    }

    if ((lane & 3) == 0) {
        const int clsl = (lane >> 2) & 3;   // class within tile
        #pragma unroll
        for (int i = 0; i < 8; ++i) {
            long r = (rowBase + i + 8 * hi) * NUM_CLASSES;
            __builtin_nontemporal_store(1.0f - m0[i], out + r + clsl);      // cls 0..3
            __builtin_nontemporal_store(1.0f - m1[i], out + r + 4 + clsl);  // cls 4..7
            if (clsl < 2)
                __builtin_nontemporal_store(1.0f - m2[i], out + r + 8 + clsl); // cls 8..9
        }
    }
}

// ------------------------------ launcher -----------------------------------
extern "C" void kernel_launch(void* const* d_in, const int* in_sizes, int n_in,
                              void* d_out, int out_size, void* d_ws, size_t ws_size,
                              hipStream_t stream) {
    const float* codes = (const float*)d_in[0];
    const float* cents = (const float*)d_in[1];
    float* out = (float*)d_out;
    __bf16* frags = (__bf16*)d_ws;   // 24 KB used

    const int batch = in_sizes[0] / CODE_DIM;        // 1048576

    hipLaunchKernelGGL(prep_centroid_frags, dim3(1), dim3(256), 0, stream,
                       cents, frags);

    // 8 waves/block * 16 rows/wave = 128 rows per block.
    const int blocks = batch / 128;
    hipLaunchKernelGGL(cosine_dist_kernel, dim3(blocks), dim3(256), 0, stream,
                       codes, frags, out);
}
